// OwnBiLSTM_18141941859055
// MI455X (gfx1250) — matrix operand order repaired and checked by reading, not compiled, not run on hardware
//
#include <hip/hip_runtime.h>
#include <stdint.h>

#define B   32
#define SQ  512
#define E   512
#define H   512
#define G   (4*H)       // 2048 gate columns
#define NWG 16          // workgroups per direction in the scan
#define NCOL (H/NWG)    // 32 h-columns per workgroup
#define LPAD 520        // padded LDS row stride (elements): 512 + 8 -> 4-dword bank skew

typedef __bf16 bf16_t;
typedef bf16_t v16bf __attribute__((ext_vector_type(16)));
typedef float  v8f   __attribute__((ext_vector_type(8)));

union BF16x16 { v16bf v; uint4 q[2]; };

// CDNA5 async global->LDS path (guarded; falls back to plain loads if absent)
#if defined(__has_builtin)
#if __has_builtin(__builtin_amdgcn_global_load_async_to_lds_b128) && __has_builtin(__builtin_amdgcn_s_wait_asynccnt)
#define USE_ASYNC_LDS 1
#endif
#endif
#ifndef USE_ASYNC_LDS
#define USE_ASYNC_LDS 0
#endif

#if USE_ASYNC_LDS
typedef int v4i __attribute__((vector_size(16)));
typedef __attribute__((address_space(1))) v4i* gv4i_p;   // global (prints as __device__)
typedef __attribute__((address_space(3))) v4i* lv4i_p;   // LDS
#endif

__device__ __forceinline__ float sigf(float x) { return 1.0f / (1.0f + __expf(-x)); }

// ---------------- utility kernels ----------------

__global__ void cast_f32_bf16(const float* __restrict__ src, bf16_t* __restrict__ dst, int n) {
  int i = blockIdx.x * 256 + threadIdx.x;
  if (i < n) dst[i] = (bf16_t)src[i];
}

__global__ void bias_sum(const float* __restrict__ a, const float* __restrict__ b,
                         float* __restrict__ out, int n) {
  int i = blockIdx.x * 256 + threadIdx.x;
  if (i < n) out[i] = a[i] + b[i];
}

__global__ void zero_u32(uint32_t* __restrict__ p, int n) {
  int i = blockIdx.x * 256 + threadIdx.x;
  if (i < n) p[i] = 0u;
}

// embedding gather + f32 -> bf16
__global__ void embed_gather(const int* __restrict__ x, const float* __restrict__ emb,
                             bf16_t* __restrict__ out) {
  int i = blockIdx.x * 256 + threadIdx.x;   // over B*S*E (exact grid)
  int tok = i / E;
  int e   = i - tok * E;
  out[i] = (bf16_t)emb[(size_t)x[tok] * E + e];
}

// ---------------- input-projection GEMM: xp = seq @ Wih^T + (bih+bhh) ----------------
// seq:[M=B*S, K] bf16 row-major, Wih:[G, K] bf16 row-major, xp:[M, G] bf16.
// Block = 256 thr (8 waves); block tile 128(M) x 64(N); wave w -> M-tile w, 4 N-tiles.
__global__ void __launch_bounds__(256) proj_gemm(
    const bf16_t* __restrict__ Aseq, const bf16_t* __restrict__ Wih,
    const float* __restrict__ bsum, bf16_t* __restrict__ xp, int K)
{
  const int wave = threadIdx.x >> 5;
  const int lane = threadIdx.x & 31;
  const int lh = lane >> 4, ll = lane & 15;
  const int m0 = blockIdx.y * 128 + wave * 16;
  const int n0 = blockIdx.x * 64;

  v8f acc[4] = {};
  const bf16_t* arowp = Aseq + (size_t)(m0 + ll) * K;
  for (int kk = 0; kk < K; kk += 32) {
    BF16x16 a;
    const bf16_t* ap = arowp + kk + lh * 8;          // A: two 16B chunks (K and K+16)
    a.q[0] = *(const uint4*)(ap);
    a.q[1] = *(const uint4*)(ap + 16);
#pragma unroll
    for (int nt = 0; nt < 4; ++nt) {
      BF16x16 bm;                                     // B: two 16B chunks (K and K+16) of Wih row
      const bf16_t* bp = Wih + (size_t)(n0 + nt * 16 + ll) * K + kk + lh * 8;
      bm.q[0] = *(const uint4*)(bp);
      bm.q[1] = *(const uint4*)(bp + 16);
      acc[nt] = __builtin_amdgcn_wmma_f32_16x16x32_bf16(
          false, a.v, false, bm.v, (short)0, acc[nt], false, false);
    }
  }
#pragma unroll
  for (int nt = 0; nt < 4; ++nt) {
    int n = n0 + nt * 16 + ll;
    float bb = bsum[n];
#pragma unroll
    for (int r = 0; r < 8; ++r) {
      int m = m0 + r + lh * 8;
      xp[(size_t)m * G + n] = (bf16_t)(acc[nt][r] + bb);
    }
  }
}

// ---------------- sequential bidirectional LSTM scan ----------------
// grid = 32 WGs: blockIdx.x/16 = direction (0 fwd, 1 bwd), %16 = column-slice.
// Each WG owns NCOL=32 h-columns -> 128 gate cols; its Whh slice (128x512 bf16,
// padded rows) is preloaded into LDS once and stays resident for all 512 steps.
// Wave w (of 8): gate q=w/2 (i,f,g,o), M-tile mt=w%2, 2 N-tiles -> 2 acc tiles.
__global__ void __launch_bounds__(256) lstm_scan(
    const bf16_t* __restrict__ xp_f, const bf16_t* __restrict__ xp_b,   // [B*S, G]
    const bf16_t* __restrict__ Whh_f, const bf16_t* __restrict__ Whh_b, // [G, H]
    bf16_t* __restrict__ hbuf,          // [2 dir][2 parity][B][H] bf16
    bf16_t* __restrict__ out_seq,       // layer0: [B,S,2H]; layer1: nullptr
    float* __restrict__ pooled,         // layer1: [B,2H]; layer0: nullptr
    const int* __restrict__ mask,       // [B,S]
    unsigned* __restrict__ bars)        // [2] per-direction step counters
{
  __shared__ bf16_t lds_w[4 * NCOL * LPAD];   // ~130 KB: resident Whh slice (gate-major)
  __shared__ bf16_t lds_h[B * LPAD];          // ~32.5 KB: previous h (padded rows)
  __shared__ float  lds_g[4 * B * NCOL];      // 16 KB: gate pre-activations

  const int dir = blockIdx.x >> 4;
  const int wg  = blockIdx.x & 15;
  const int j0  = wg * NCOL;
  const int tid = threadIdx.x;
  const int wave = tid >> 5, lane = tid & 31;
  const int q  = wave >> 1;
  const int mt = wave & 1;
  const int lh = lane >> 4, ll = lane & 15;

  const bf16_t* xp  = dir ? xp_b  : xp_f;
  const bf16_t* Whh = dir ? Whh_b : Whh_f;
  unsigned* bar = bars + dir;

  // one-time preload of this WG's Whh slice: local row rl = q*NCOL + j  <->  global row q*H + j0 + j
#pragma unroll 4
  for (int idx = tid; idx < 4 * NCOL * (H / 8); idx += 256) {   // 64 uint4 per row
    int rl = idx >> 6, c = idx & 63;
    const uint4* src = (const uint4*)(Whh + (size_t)((rl >> 5) * H + j0 + (rl & 31)) * H) + c;
    ((uint4*)lds_w)[rl * (LPAD / 8) + c] = *src;
  }

  // gating-phase ownership: thread -> (batch row gb, 4 cols at gj0); step-invariant
  const int gb  = tid >> 3;
  const int gj0 = (tid & 7) * 4;
  float creg[4] = {};     // cell state, registers for the whole scan
  float preg[4] = {};     // masked pooled sum

  for (int step = 0; step < SQ; ++step) {
    const int t = dir ? (SQ - 1 - step) : step;
    const int par_r = step & 1;
    const int par_w = par_r ^ 1;

    // stage h(t-1) (all 512 cols) into padded LDS rows
    {
      const bf16_t* src = hbuf + (size_t)(dir * 2 + par_r) * B * H;
#if USE_ASYNC_LDS
#pragma unroll
      for (int i = 0; i < 8; ++i) {
        int e = (tid + i * 256) * 8;              // element index in [0, B*H)
        int row = e >> 9, col = e & 511;
        __builtin_amdgcn_global_load_async_to_lds_b128(
            (gv4i_p)(uintptr_t)(src + (size_t)row * H + col),
            (lv4i_p)(uintptr_t)(lds_h + row * LPAD + col),
            0, 0);
      }
      __builtin_amdgcn_s_wait_asynccnt(0);
#else
#pragma unroll
      for (int i = 0; i < 8; ++i) {
        int idx = tid + i * 256;
        int row = idx >> 6, c = idx & 63;
        ((uint4*)lds_h)[row * (LPAD / 8) + c] = ((const uint4*)src)[idx];
      }
#endif
    }
    __syncthreads();

    // h(t-1) @ Whh^T : LDS-only operands
    v8f acc[2] = {};
#pragma unroll 4
    for (int kk = 0; kk < H; kk += 32) {
      BF16x16 a;
      const bf16_t* ap = lds_h + (mt * 16 + ll) * LPAD + kk + lh * 8;
      a.q[0] = *(const uint4*)(ap);
      a.q[1] = *(const uint4*)(ap + 16);
#pragma unroll
      for (int nt = 0; nt < 2; ++nt) {
        BF16x16 bm;
        const bf16_t* bp = lds_w + (q * NCOL + nt * 16 + ll) * LPAD + kk + lh * 8;
        bm.q[0] = *(const uint4*)(bp);
        bm.q[1] = *(const uint4*)(bp + 16);
        acc[nt] = __builtin_amdgcn_wmma_f32_16x16x32_bf16(
            false, a.v, false, bm.v, (short)0, acc[nt], false, false);
      }
    }
    // scatter gate pre-activations: lds_g[q][m][jl]
#pragma unroll
    for (int nt = 0; nt < 2; ++nt) {
      int jl = nt * 16 + ll;
#pragma unroll
      for (int r = 0; r < 8; ++r) {
        int m = mt * 16 + r + lh * 8;
        lds_g[(q * B + m) * NCOL + jl] = acc[nt][r];
      }
    }
    __syncthreads();

    // elementwise gating for (gb, gj0..gj0+3)
    const bf16_t* xrow = xp + (size_t)(gb * SQ + t) * G + j0 + gj0;
    const float* gi = lds_g + (0 * B + gb) * NCOL + gj0;
    const float* gf = lds_g + (1 * B + gb) * NCOL + gj0;
    const float* gg = lds_g + (2 * B + gb) * NCOL + gj0;
    const float* go = lds_g + (3 * B + gb) * NCOL + gj0;
    const float mval = (float)mask[gb * SQ + t];

    union { bf16_t h[4]; uint2 u; } ho;
#pragma unroll
    for (int u = 0; u < 4; ++u) {
      float iv = sigf(gi[u] + (float)xrow[0 * H + u]);
      float fv = sigf(gf[u] + (float)xrow[1 * H + u]);
      float gv = tanhf(gg[u] + (float)xrow[2 * H + u]);
      float ov = sigf(go[u] + (float)xrow[3 * H + u]);
      float c = fv * creg[u] + iv * gv;
      creg[u] = c;
      float h = ov * tanhf(c);
      preg[u] += h * mval;
      ho.h[u] = (bf16_t)h;
    }
    // publish new h (double buffered) + layer output
    *(uint2*)(hbuf + (size_t)(dir * 2 + par_w) * B * H + gb * H + j0 + gj0) = ho.u;
    if (out_seq)
      *(uint2*)(out_seq + (size_t)(gb * SQ + t) * (2 * H) + dir * H + j0 + gj0) = ho.u;

    // prefetch next step's xp segments into cache (global_prefetch_b8)
    if (step + 1 < SQ) {
      const int tn = dir ? (t - 1) : (t + 1);
      const bf16_t* xn = xp + (size_t)(gb * SQ + tn) * G + j0 + gj0;
      __builtin_prefetch(xn + 0 * H, 0, 0);
      __builtin_prefetch(xn + 1 * H, 0, 0);
      __builtin_prefetch(xn + 2 * H, 0, 0);
      __builtin_prefetch(xn + 3 * H, 0, 0);
    }

    // grid barrier across the 16 WGs of this direction
    __syncthreads();
    if (tid == 0) {
      __threadfence();
      atomicAdd(bar, 1u);
      const unsigned target = (unsigned)(step + 1) * NWG;
      while (atomicAdd(bar, 0u) < target) { __builtin_amdgcn_s_sleep(2); }
      __threadfence();
    }
    __syncthreads();
  }

  if (pooled) {
    float* pd = pooled + gb * (2 * H) + dir * H + j0 + gj0;
#pragma unroll
    for (int u = 0; u < 4; ++u) pd[u] = preg[u];
  }
}

// ---------------- classifier head ----------------
__global__ void __launch_bounds__(128) head_kernel(
    const float* __restrict__ pooled, const int* __restrict__ mask,
    const float* __restrict__ fcW, const float* __restrict__ fcb,
    float* __restrict__ out)
{
  __shared__ float red[128];
  const int b = blockIdx.x, tid = threadIdx.x;

  float ms = 0.f;
  for (int i = tid; i < SQ; i += 128) ms += (float)mask[b * SQ + i];
  red[tid] = ms; __syncthreads();
  for (int s2 = 64; s2 > 0; s2 >>= 1) { if (tid < s2) red[tid] += red[tid + s2]; __syncthreads(); }
  const float inv = 1.0f / fmaxf(red[0], 1e-9f);
  __syncthreads();

  for (int c = 0; c < 2; ++c) {
    float acc = 0.f;
    for (int k = tid; k < 2 * H; k += 128)
      acc += pooled[b * 2 * H + k] * fcW[c * 2 * H + k];
    red[tid] = acc; __syncthreads();
    for (int s2 = 64; s2 > 0; s2 >>= 1) { if (tid < s2) red[tid] += red[tid + s2]; __syncthreads(); }
    if (tid == 0) out[b * 2 + c] = red[0] * inv + fcb[c];
    __syncthreads();
  }
}

// ---------------- host side ----------------
extern "C" void kernel_launch(void* const* d_in, const int* in_sizes, int n_in,
                              void* d_out, int out_size, void* d_ws, size_t ws_size,
                              hipStream_t stream) {
  const int*   x    = (const int*)d_in[0];
  const int*   amask= (const int*)d_in[1];
  const float* emb  = (const float*)d_in[2];
  const float* fcW  = (const float*)d_in[3];
  const float* fcb  = (const float*)d_in[4];
  // weights: [5..8]=fwd l0, [9..12]=bwd l0, [13..16]=fwd l1, [17..20]=bwd l1
  const float* WihF0 = (const float*)d_in[5];  const float* WhhF0 = (const float*)d_in[6];
  const float* bihF0 = (const float*)d_in[7];  const float* bhhF0 = (const float*)d_in[8];
  const float* WihB0 = (const float*)d_in[9];  const float* WhhB0 = (const float*)d_in[10];
  const float* bihB0 = (const float*)d_in[11]; const float* bhhB0 = (const float*)d_in[12];
  const float* WihF1 = (const float*)d_in[13]; const float* WhhF1 = (const float*)d_in[14];
  const float* bihF1 = (const float*)d_in[15]; const float* bhhF1 = (const float*)d_in[16];
  const float* WihB1 = (const float*)d_in[17]; const float* WhhB1 = (const float*)d_in[18];
  const float* bihB1 = (const float*)d_in[19]; const float* bhhB1 = (const float*)d_in[20];

  // workspace carve-out
  char* ws = (char*)d_ws;
  size_t off = 0;
  auto take = [&](size_t bytes) { size_t o = off; off = (off + bytes + 255) & ~(size_t)255; return (void*)(ws + o); };

  bf16_t* wihF0 = (bf16_t*)take((size_t)G * E * 2);
  bf16_t* wihB0 = (bf16_t*)take((size_t)G * E * 2);
  bf16_t* wihF1 = (bf16_t*)take((size_t)G * 2 * H * 2);
  bf16_t* wihB1 = (bf16_t*)take((size_t)G * 2 * H * 2);
  bf16_t* whhF0 = (bf16_t*)take((size_t)G * H * 2);
  bf16_t* whhB0 = (bf16_t*)take((size_t)G * H * 2);
  bf16_t* whhF1 = (bf16_t*)take((size_t)G * H * 2);
  bf16_t* whhB1 = (bf16_t*)take((size_t)G * H * 2);
  float* bsF0 = (float*)take((size_t)G * 4);
  float* bsB0 = (float*)take((size_t)G * 4);
  float* bsF1 = (float*)take((size_t)G * 4);
  float* bsB1 = (float*)take((size_t)G * 4);
  bf16_t* seq0 = (bf16_t*)take((size_t)B * SQ * E * 2);
  bf16_t* seq1 = (bf16_t*)take((size_t)B * SQ * 2 * H * 2);
  bf16_t* xpF  = (bf16_t*)take((size_t)B * SQ * G * 2);
  bf16_t* xpB  = (bf16_t*)take((size_t)B * SQ * G * 2);
  bf16_t* hbuf = (bf16_t*)take((size_t)2 * 2 * B * H * 2);
  float* pooled = (float*)take((size_t)B * 2 * H * 4);
  unsigned* bars = (unsigned*)take(4 * sizeof(unsigned));
  (void)ws_size; (void)n_in; (void)in_sizes; (void)out_size;

  // weight / bias prep
  const int nW0 = G * E, nW1 = G * 2 * H, nWh = G * H;
  cast_f32_bf16<<<(nW0 + 255) / 256, 256, 0, stream>>>(WihF0, wihF0, nW0);
  cast_f32_bf16<<<(nW0 + 255) / 256, 256, 0, stream>>>(WihB0, wihB0, nW0);
  cast_f32_bf16<<<(nW1 + 255) / 256, 256, 0, stream>>>(WihF1, wihF1, nW1);
  cast_f32_bf16<<<(nW1 + 255) / 256, 256, 0, stream>>>(WihB1, wihB1, nW1);
  cast_f32_bf16<<<(nWh + 255) / 256, 256, 0, stream>>>(WhhF0, whhF0, nWh);
  cast_f32_bf16<<<(nWh + 255) / 256, 256, 0, stream>>>(WhhB0, whhB0, nWh);
  cast_f32_bf16<<<(nWh + 255) / 256, 256, 0, stream>>>(WhhF1, whhF1, nWh);
  cast_f32_bf16<<<(nWh + 255) / 256, 256, 0, stream>>>(WhhB1, whhB1, nWh);
  bias_sum<<<G / 256, 256, 0, stream>>>(bihF0, bhhF0, bsF0, G);
  bias_sum<<<G / 256, 256, 0, stream>>>(bihB0, bhhB0, bsB0, G);
  bias_sum<<<G / 256, 256, 0, stream>>>(bihF1, bhhF1, bsF1, G);
  bias_sum<<<G / 256, 256, 0, stream>>>(bihB1, bhhB1, bsB1, G);

  // embedding gather -> bf16
  embed_gather<<<(B * SQ * E) / 256, 256, 0, stream>>>(x, emb, seq0);

  // zero h state + step barriers (deterministic per call)
  zero_u32<<<((2 * 2 * B * H * 2) / 4 + 255) / 256, 256, 0, stream>>>((uint32_t*)hbuf, (2 * 2 * B * H * 2) / 4);
  zero_u32<<<1, 256, 0, stream>>>((uint32_t*)bars, 4);

  const dim3 pgrid(G / 64, (B * SQ) / 128);
  // ---- layer 0 ----
  proj_gemm<<<pgrid, 256, 0, stream>>>(seq0, wihF0, bsF0, xpF, E);
  proj_gemm<<<pgrid, 256, 0, stream>>>(seq0, wihB0, bsB0, xpB, E);
  lstm_scan<<<2 * NWG, 256, 0, stream>>>(xpF, xpB, whhF0, whhB0, hbuf,
                                         seq1, (float*)nullptr, amask, bars);
  // ---- layer 1 ----
  zero_u32<<<((2 * 2 * B * H * 2) / 4 + 255) / 256, 256, 0, stream>>>((uint32_t*)hbuf, (2 * 2 * B * H * 2) / 4);
  proj_gemm<<<pgrid, 256, 0, stream>>>(seq1, wihF1, bsF1, xpF, 2 * H);
  proj_gemm<<<pgrid, 256, 0, stream>>>(seq1, wihB1, bsB1, xpB, 2 * H);
  lstm_scan<<<2 * NWG, 256, 0, stream>>>(xpF, xpB, whhF1, whhB1, hbuf,
                                         (bf16_t*)nullptr, pooled, amask, bars + 2);
  // ---- head ----
  head_kernel<<<B, 128, 0, stream>>>(pooled, amask, fcW, fcb, (float*)d_out);
}
